// MERG_2989297238264
// MI455X (gfx1250) — compile-verified, matrix operand order checked by reading, not run on hardware
//
#include <hip/hip_runtime.h>
#include <hip/hip_bf16.h>

// ---------------------------------------------------------------------------
// MI455X (gfx1250) implementation. wave32, WMMA bf16 16x16x32, fp32 accum.
// Weights are one-time packed to bf16 [N][K] so GEMM staging is pure b128.
// ---------------------------------------------------------------------------

typedef __attribute__((ext_vector_type(16))) __bf16 v16bf;
typedef __attribute__((ext_vector_type(8)))  __bf16 v8bf;
typedef __attribute__((ext_vector_type(8)))  float  v8f;

#define NB_   32
#define NN_   64
#define EPG_  2048
#define ETOT_ 65536
#define NTOT_ 2048   // NB_*NN_
#define D_    256

__device__ inline float waveRed(float v) {
  #pragma unroll
  for (int o = 16; o; o >>= 1) v += __shfl_xor(v, o, 32);
  return v;
}

// One-time: pack f32 row-major [K][N] weight into bf16 [N][K] (transposed).
__global__ __launch_bounds__(256) void packW(
    const float* __restrict__ W, __bf16* __restrict__ Wt, int K, int N)
{
  const int i = blockIdx.x * 256 + threadIdx.x;
  if (i >= K * N) return;
  const int k = i / N, n = i - k * N;
  Wt[(long long)n * K + k] = (__bf16)W[i];
}

// ---------------------------------------------------------------------------
// Generic tiled WMMA GEMM:  C = act( rowScale*(A@W) + bias + resid )
//   A: [M,K] f32. Wt: bf16 [N][ldw] pre-transposed (element (n,k) at n*ldw+k).
//   Block tile 64x64, 8 waves (4 Mtiles x 2 Ntiles of 16x32 each).
//   blockIdx.z = batch; aB/wB/cB/rsB are element strides per batch.
//   act: 0=none, 1=relu, 2=elu(x)+1
// ---------------------------------------------------------------------------
__global__ __launch_bounds__(256) void gemm_wmma(
    const float* __restrict__ A, const __bf16* __restrict__ Wg,
    const float* __restrict__ bias, const float* __restrict__ resid,
    const float* __restrict__ rowScale, float* __restrict__ C,
    int M, int N, int K, int ldw,
    long long aB, long long wB, long long cB, long long rsB,
    int act, float rsMul)
{
  __shared__ __attribute__((aligned(16))) __bf16 As[64][32];   // [m][k]
  __shared__ __attribute__((aligned(16))) __bf16 Ws[64][32];   // [n][k]

  const int bz = blockIdx.z;
  A  += (long long)bz * aB;
  Wg += (long long)bz * wB;
  C  += (long long)bz * cB;
  if (resid)    resid    += (long long)bz * cB;
  if (rowScale) rowScale += (long long)bz * rsB;

  const int m0   = blockIdx.x * 64;
  const int n0   = blockIdx.y * 64;
  const int tid  = threadIdx.x;
  const int lane = tid & 31;
  const int wv   = tid >> 5;    // 0..7
  const int mt   = wv >> 1;     // 0..3
  const int nt   = wv & 1;      // 0..1

  v8f acc0 = {};
  v8f acc1 = {};

  const int ar = tid >> 2;         // 0..63  A row in tile
  const int ac = (tid & 3) * 8;    // 0,8,16,24
  const int wn = tid >> 2;         // 0..63  W n-row in tile
  const int wk = (tid & 3) * 8;    // 0,8,16,24

  const int gmA = m0 + ar;
  const float* aRow = A + (long long)gmA * K;

  for (int k0 = 0; k0 < K; k0 += 32) {
    // ---- stage A tile: two coalesced b128 loads, pk cvt, one b128 LDS store
    {
      v8bf ap = {};
      if (gmA < M) {
        const float4* s4 = (const float4*)(aRow + k0 + ac);
        const float4 f0 = s4[0];
        const float4 f1 = s4[1];
        ap[0] = (__bf16)f0.x; ap[1] = (__bf16)f0.y;
        ap[2] = (__bf16)f0.z; ap[3] = (__bf16)f0.w;
        ap[4] = (__bf16)f1.x; ap[5] = (__bf16)f1.y;
        ap[6] = (__bf16)f1.z; ap[7] = (__bf16)f1.w;
        if ((k0 + 32) < K) __builtin_prefetch((const float*)s4 + 32, 0, 0);
      }
      *(v8bf*)&As[ar][ac] = ap;
    }
    // ---- stage W tile: pre-packed bf16, straight b128 copy ----
    {
      const v8bf w8 = *(const v8bf*)(Wg + (long long)(n0 + wn) * ldw + k0 + wk);
      *(v8bf*)&Ws[wn][wk] = w8;
    }
    __syncthreads();

    // ---- fragments per ISA 16-bit 16x32 layout (lane half selects K group)
    const int half  = lane >> 4;
    const int mrow  = (lane & 15) + mt * 16;
    const int nrow0 = (lane & 15) + nt * 32;

    v8bf alo  = *(const v8bf*)&As[mrow][half * 8];
    v8bf ahi  = *(const v8bf*)&As[mrow][16 + half * 8];
    v8bf b0lo = *(const v8bf*)&Ws[nrow0][half * 8];
    v8bf b0hi = *(const v8bf*)&Ws[nrow0][16 + half * 8];
    v8bf b1lo = *(const v8bf*)&Ws[nrow0 + 16][half * 8];
    v8bf b1hi = *(const v8bf*)&Ws[nrow0 + 16][16 + half * 8];

    v16bf a, b0, b1;
    #pragma unroll
    for (int i = 0; i < 8; ++i) {
      a[i]  = alo[i];  a[i + 8]  = ahi[i];
      b0[i] = b0lo[i]; b0[i + 8] = b0hi[i];
      b1[i] = b1lo[i]; b1[i + 8] = b1hi[i];
    }
    acc0 = __builtin_amdgcn_wmma_f32_16x16x32_bf16(false, a, false, b0, (short)0, acc0, false, false);
    acc1 = __builtin_amdgcn_wmma_f32_16x16x32_bf16(false, a, false, b1, (short)0, acc1, false, false);
    __syncthreads();
  }

  // ---- epilogue (C/D layout: lane n = lane&15, row = (lane>>4)*8 + r) ----
  const int nlo = lane & 15;
  const int mb  = mt * 16 + (lane >> 4) * 8;
  #pragma unroll
  for (int r = 0; r < 8; ++r) {
    const int gm = m0 + mb + r;
    if (gm >= M) continue;
    const float rs = rowScale ? rowScale[gm] * rsMul : 1.0f;
    const int c0 = n0 + nt * 32 + nlo;
    const int c1 = c0 + 16;
    float v0 = acc0[r] * rs;
    float v1 = acc1[r] * rs;
    if (bias)  { v0 += bias[c0]; v1 += bias[c1]; }
    if (resid) { v0 += resid[(long long)gm * N + c0]; v1 += resid[(long long)gm * N + c1]; }
    if (act == 1)      { v0 = fmaxf(v0, 0.0f); v1 = fmaxf(v1, 0.0f); }
    else if (act == 2) { v0 = (v0 > 0.0f) ? v0 + 1.0f : __expf(v0);
                         v1 = (v1 > 0.0f) ? v1 + 1.0f : __expf(v1); }
    C[(long long)gm * N + c0] = v0;
    C[(long long)gm * N + c1] = v1;
  }
}

// ---------------------------------------------------------------------------
// Batched KV:  KVt[b][v][d] = (1/64) * sum_i elu1(k2[b,i,d]) * v2[b,i,v]
// (bf16, pre-transposed for the following Q@KV WMMA GEMM)
// Also Ksum[b,d] = 64 * sum_i elu1(k2[b,i,d]).  One block per graph.
// ---------------------------------------------------------------------------
__global__ __launch_bounds__(256) void kv_wmma(
    const float* __restrict__ k2, const float* __restrict__ v2,
    __bf16* __restrict__ KVt, float* __restrict__ Ksum)
{
  extern __shared__ __attribute__((aligned(16))) char kvsm[];
  __bf16* K2t = (__bf16*)kvsm;      // [256][64]  K2t[d][i] = elu1(k2[b,i,d])
  __bf16* V2t = K2t + 256 * 64;     // [256][64]  V2t[v][i] = v2[b,i,v]/64

  const int b   = blockIdx.x;
  const int tid = threadIdx.x;
  const float* kb = k2 + (long long)b * 64 * 256;
  const float* vb = v2 + (long long)b * 64 * 256;

  for (int idx = tid; idx < 64 * 256; idx += 256) {
    const int i = idx >> 8, d = idx & 255;
    const float kv = kb[idx];
    K2t[d * 64 + i] = (__bf16)((kv > 0.0f) ? kv + 1.0f : __expf(kv));
    V2t[d * 64 + i] = (__bf16)(vb[idx] * (1.0f / 64.0f));
  }
  __syncthreads();

  { // Ksum over all 4096 virtual source rows = 64 * sum_i
    float s = 0.0f;
    #pragma unroll 8
    for (int i = 0; i < 64; ++i) s += (float)K2t[tid * 64 + i];
    Ksum[b * 256 + tid] = s * 64.0f;
  }

  const int lane = tid & 31, wv = tid >> 5;
  const int half = lane >> 4;
  for (int t = wv; t < 256; t += 8) {
    const int mt = t >> 4, ntile = t & 15;
    const int mrow = mt * 16 + (lane & 15);
    const int nrow = ntile * 16 + (lane & 15);
    v8f acc = {};
    #pragma unroll
    for (int k0 = 0; k0 < 64; k0 += 32) {
      v8bf alo = *(const v8bf*)&K2t[mrow * 64 + k0 + half * 8];
      v8bf ahi = *(const v8bf*)&K2t[mrow * 64 + k0 + 16 + half * 8];
      v8bf blo = *(const v8bf*)&V2t[nrow * 64 + k0 + half * 8];
      v8bf bhi = *(const v8bf*)&V2t[nrow * 64 + k0 + 16 + half * 8];
      v16bf a, bb;
      #pragma unroll
      for (int i = 0; i < 8; ++i) {
        a[i] = alo[i];  a[i + 8] = ahi[i];
        bb[i] = blo[i]; bb[i + 8] = bhi[i];
      }
      acc = __builtin_amdgcn_wmma_f32_16x16x32_bf16(false, a, false, bb, (short)0, acc, false, false);
    }
    // Store transposed as bf16: KVt[b][n][mb..mb+7] (one b128 store)
    const int mb = mt * 16 + (lane >> 4) * 8;
    const int nn = ntile * 16 + (lane & 15);
    v8bf o;
    #pragma unroll
    for (int r = 0; r < 8; ++r) o[r] = (__bf16)acc[r];
    *(v8bf*)&KVt[((long long)b * 256 + nn) * 256 + mb] = o;
  }
}

// ---------------------------------------------------------------------------
// Edge "conv": cross-correlation width-3 over feature axis of (src,dst) pair.
// ---------------------------------------------------------------------------
__global__ __launch_bounds__(256) void conv_edge(
    const float* __restrict__ h, const int* __restrict__ esrc,
    const int* __restrict__ edst, const float* __restrict__ cw,
    const float* __restrict__ cb, float* __restrict__ out)
{
  const int eid = blockIdx.x;
  const int f   = threadIdx.x;
  const int b   = eid >> 11;
  const int gs  = (b << 6) + esrc[eid];
  const int gd  = (b << 6) + edst[eid];
  const float* r0 = h + (long long)gs * 256;
  const float* r1 = h + (long long)gd * 256;
  float acc = cb[0];
  const float l0 = (f > 0)   ? r0[f - 1] : 0.0f;
  const float u0 = (f < 255) ? r0[f + 1] : 0.0f;
  const float l1 = (f > 0)   ? r1[f - 1] : 0.0f;
  const float u1 = (f < 255) ? r1[f + 1] : 0.0f;
  acc += l0 * cw[0] + r0[f] * cw[1] + u0 * cw[2];
  acc += l1 * cw[3] + r1[f] * cw[4] + u1 * cw[5];
  out[(long long)eid * 256 + f] = acc;
}

// ---------------------------------------------------------------------------
// GatedGCN edge stage: per-graph LDS segment-sum with ds_add_f32 atomics.
// ---------------------------------------------------------------------------
__global__ __launch_bounds__(1024) void gcn_edge(
    const int* __restrict__ esrc, const int* __restrict__ edst,
    const float* __restrict__ Dhh, const float* __restrict__ Ehh,
    const float* __restrict__ Ce, const float* __restrict__ Bhh,
    float* __restrict__ numO, float* __restrict__ denO)
{
  extern __shared__ float gsm[];
  float* numS = gsm;              // [64][256]
  float* denS = gsm + 64 * 256;   // [64][256]
  const int b   = blockIdx.x;
  const int tid = threadIdx.x;
  for (int i = tid; i < 64 * 256; i += 1024) { numS[i] = 0.0f; denS[i] = 0.0f; }
  __syncthreads();

  const int er = tid >> 8;    // 4 edges in flight
  const int f  = tid & 255;
  for (int e0 = 0; e0 < 2048; e0 += 4) {
    const int eL  = e0 + er;
    const int src = esrc[b * 2048 + eL];
    const int dst = edst[b * 2048 + eL];
    const int gs  = b * 64 + src;
    const int gd  = b * 64 + dst;
    const float x = Dhh[gs * 256 + f] + Ehh[gd * 256 + f]
                  + Ce[((long long)(b * 2048 + eL)) * 256 + f];
    const float sg = 1.0f / (1.0f + __expf(-x));
    atomicAdd(&numS[dst * 256 + f], sg * Bhh[gs * 256 + f]);
    atomicAdd(&denS[dst * 256 + f], sg);
  }
  __syncthreads();
  for (int i = tid; i < 64 * 256; i += 1024) {
    numO[(long long)b * 64 * 256 + i] = numS[i];
    denO[(long long)b * 64 * 256 + i] = denS[i];
  }
}

__global__ void gcn_node(const float* __restrict__ Ah, const float* __restrict__ num,
                         const float* __restrict__ den, float* __restrict__ hn,
                         long long n)
{
  long long i = (long long)blockIdx.x * blockDim.x + threadIdx.x;
  const long long st = (long long)gridDim.x * blockDim.x;
  for (; i < n; i += st) hn[i] = Ah[i] + num[i] / (den[i] + 1e-6f);
}

// ---------------------------------------------------------------------------
// Column mean/var (BatchNorm training stats), two-stage.
// ---------------------------------------------------------------------------
__global__ __launch_bounds__(256) void colstat_part(
    const float* __restrict__ x, int M, float* __restrict__ part)
{
  const int f  = threadIdx.x;
  const int r0 = blockIdx.x * 256;
  float s = 0.0f, sq = 0.0f;
  for (int r = r0; r < r0 + 256 && r < M; ++r) {
    const float v = x[(long long)r * 256 + f];
    s += v; sq += v * v;
  }
  part[blockIdx.x * 512 + f]       = s;
  part[blockIdx.x * 512 + 256 + f] = sq;
}

__global__ __launch_bounds__(256) void colstat_fin(
    const float* __restrict__ part, int nb, int M,
    float* __restrict__ mean, float* __restrict__ var)
{
  const int f = threadIdx.x;
  float s = 0.0f, sq = 0.0f;
  for (int ib = 0; ib < nb; ++ib) { s += part[ib * 512 + f]; sq += part[ib * 512 + 256 + f]; }
  const float m = s / (float)M;
  mean[f] = m;
  var[f]  = sq / (float)M - m * m;
}

// out = relu((x-mean)*rsqrt(var+eps)*g + b) [+ resid]
__global__ void bn_apply(const float* __restrict__ x, const float* __restrict__ mean,
                         const float* __restrict__ var, const float* __restrict__ g,
                         const float* __restrict__ bta, const float* __restrict__ resid,
                         float* __restrict__ out, long long n)
{
  long long i = (long long)blockIdx.x * blockDim.x + threadIdx.x;
  const long long st = (long long)gridDim.x * blockDim.x;
  for (; i < n; i += st) {
    const int f = (int)(i & 255);
    float y = (x[i] - mean[f]) * rsqrtf(var[f] + 1e-5f) * g[f] + bta[f];
    y = fmaxf(y, 0.0f);
    if (resid) y += resid[i];
    out[i] = y;
  }
}

// Row LayerNorm over 256 features, one wave32 per row. out = [resid +] LN(x)
__global__ __launch_bounds__(256) void ln_kernel(
    const float* __restrict__ x, const float* __restrict__ g,
    const float* __restrict__ bta, const float* __restrict__ resid,
    float* __restrict__ out, int M)
{
  const int wv = threadIdx.x >> 5, lane = threadIdx.x & 31;
  const int row = blockIdx.x * 8 + wv;
  if (row >= M) return;
  const float* xr = x + (long long)row * 256;
  float v[8]; float s = 0.0f, sq = 0.0f;
  #pragma unroll
  for (int i = 0; i < 8; ++i) { v[i] = xr[lane * 8 + i]; s += v[i]; sq += v[i] * v[i]; }
  s = waveRed(s); sq = waveRed(sq);
  const float m = s / 256.0f;
  const float rstd = rsqrtf(sq / 256.0f - m * m + 1e-5f);
  float* orow = out + (long long)row * 256;
  const float* rr = resid ? resid + (long long)row * 256 : nullptr;
  #pragma unroll
  for (int i = 0; i < 8; ++i) {
    const int c = lane * 8 + i;
    float y = (v[i] - m) * rstd * g[c] + bta[c];
    if (rr) y += rr[c];
    orow[c] = y;
  }
}

__global__ __launch_bounds__(256) void mean_nodes(
    const float* __restrict__ x, float* __restrict__ gg)
{
  const int b = blockIdx.x, f = threadIdx.x;
  float s = 0.0f;
  for (int j = 0; j < 64; ++j) s += x[((long long)(b * 64 + j)) * 256 + f];
  gg[b * 256 + f] = s * (1.0f / 64.0f);
}

// Encoder-1 linear attention (source rows are one broadcast row per graph):
// msg[row] = vg[b] * (64*<Q,Kg>)/(64*<Q,Kg>+1e-6)
__global__ __launch_bounds__(256) void attn1_kernel(
    const float* __restrict__ Q, const float* __restrict__ Kg,
    const float* __restrict__ vg, float* __restrict__ msg, int M)
{
  const int wv = threadIdx.x >> 5, lane = threadIdx.x & 31;
  const int row = blockIdx.x * 8 + wv;
  if (row >= M) return;
  const int b = row >> 6;
  const float* qr = Q + (long long)row * 256;
  const float* kr = Kg + b * 256;
  float s = 0.0f;
  #pragma unroll
  for (int i = 0; i < 8; ++i) s += qr[lane * 8 + i] * kr[lane * 8 + i];
  s = waveRed(s);
  const float num = 64.0f * s;
  const float fac = num / (num + 1e-6f);
  const float* vr = vg + b * 256;
  float* mr = msg + (long long)row * 256;
  #pragma unroll
  for (int i = 0; i < 8; ++i) mr[lane * 8 + i] = vr[lane * 8 + i] * fac;
}

// Z[row] = 1/(<Q2e[row], Ksum[b]> + 1e-6)
__global__ __launch_bounds__(256) void z2_kernel(
    const float* __restrict__ Q, const float* __restrict__ Ksum,
    float* __restrict__ Z, int M)
{
  const int wv = threadIdx.x >> 5, lane = threadIdx.x & 31;
  const int row = blockIdx.x * 8 + wv;
  if (row >= M) return;
  const int b = row >> 6;
  const float* qr = Q + (long long)row * 256;
  const float* kr = Ksum + b * 256;
  float s = 0.0f;
  #pragma unroll
  for (int i = 0; i < 8; ++i) s += qr[lane * 8 + i] * kr[lane * 8 + i];
  s = waveRed(s);
  if (lane == 0) Z[row] = 1.0f / (s + 1e-6f);
}

// pre[eid,f] = lr_local[eid,f] + g2[b*64 + edge_dst[eid], f]
__global__ void edge_pre(const float* __restrict__ loc, const float* __restrict__ g2,
                         const int* __restrict__ edst, float* __restrict__ pre,
                         long long n)
{
  long long i = (long long)blockIdx.x * blockDim.x + threadIdx.x;
  const long long st = (long long)gridDim.x * blockDim.x;
  for (; i < n; i += st) {
    const int eid = (int)(i >> 8);
    const int f   = (int)(i & 255);
    const int b   = eid >> 11;
    const int d   = edst[eid];
    pre[i] = loc[i] + g2[((long long)(b << 6) + d) * 256 + f];
  }
}

// ---------------------------------------------------------------------------
// Host orchestration
// ---------------------------------------------------------------------------
static inline void gemm(hipStream_t s, const float* A, const __bf16* W, int ldw,
                        const float* bias, const float* res, const float* rs,
                        float* C, int M, int N, int K, int act,
                        float rsMul = 1.0f, int batch = 1,
                        long long aB = 0, long long wB = 0,
                        long long cB = 0, long long rsB = 0)
{
  dim3 g((unsigned)((M + 63) / 64), (unsigned)(N / 64), (unsigned)batch);
  gemm_wmma<<<g, 256, 0, s>>>(A, W, bias, res, rs, C, M, N, K, ldw,
                              aB, wB, cB, rsB, act, rsMul);
}

extern "C" void kernel_launch(void* const* d_in, const int* in_sizes, int n_in,
                              void* d_out, int out_size, void* d_ws, size_t ws_size,
                              hipStream_t stream)
{
  (void)in_sizes; (void)n_in; (void)out_size; (void)ws_size;

  const float* h      = (const float*)d_in[0];
  const float* e      = (const float*)d_in[1];
  const int*   esrc   = (const int*)d_in[2];
  const int*   edst   = (const int*)d_in[3];
  const float* conv_w = (const float*)d_in[4];
  const float* conv_b = (const float*)d_in[5];
  const float* ep2_w  = (const float*)d_in[6];
  const float* ep2_b  = (const float*)d_in[7];
  const float* ep3_w  = (const float*)d_in[8];
  const float* ep3_b  = (const float*)d_in[9];
  const float* ep4_w  = (const float*)d_in[10];
  const float* ep4_b  = (const float*)d_in[11];
  const float* bn_g   = (const float*)d_in[12];
  const float* bn_b   = (const float*)d_in[13];
  const float* A_w    = (const float*)d_in[14];
  const float* A_b    = (const float*)d_in[15];
  const float* B_w    = (const float*)d_in[16];
  const float* B_b    = (const float*)d_in[17];
  const float* C_w    = (const float*)d_in[18];
  const float* C_b    = (const float*)d_in[19];
  const float* D_w    = (const float*)d_in[20];
  const float* D_b    = (const float*)d_in[21];
  const float* E_w    = (const float*)d_in[22];
  const float* E_b    = (const float*)d_in[23];
  const float* bnh_g  = (const float*)d_in[24];
  const float* bnh_b  = (const float*)d_in[25];
  const float* wq     = (const float*)d_in[26];
  const float* wk     = (const float*)d_in[27];
  const float* wvv    = (const float*)d_in[28];
  const float* wm     = (const float*)d_in[29];
  const float* w1     = (const float*)d_in[30];
  const float* w2     = (const float*)d_in[31];
  const float* ln1g   = (const float*)d_in[32];
  const float* ln1b   = (const float*)d_in[33];
  const float* ln2g   = (const float*)d_in[34];
  const float* ln2b   = (const float*)d_in[35];

  const size_t NE = (size_t)ETOT_ * 256;   // 16,777,216
  const size_t NV = (size_t)NTOT_ * 256;   // 524,288

  float* wsf = (float*)d_ws;
  size_t off = 0;
  auto take = [&](size_t n) { float* p = wsf + off; off += n; return p; };

  float* lrConv  = take(NE);
  float* lrLocal = take(NE);
  float* Ce      = take(NE);          // later reused as BN "pre" buffer
  float* hA      = take(NV);
  float* hBv     = take(NV);
  float* Ahb     = take(NV);
  float* Bhb     = take(NV);
  float* Dhb     = take(NV);
  float* Ehb     = take(NV);
  float* numb    = take(NV);
  float* denb    = take(NV);
  float* hn      = take(NV);
  float* part    = take(256 * 512);
  float* meanv   = take(256);
  float* varv    = take(256);
  float* hsb     = take(NV);
  float* gg      = take(32 * 256);
  float* Kgb     = take(32 * 256);
  float* vgb     = take(32 * 256);
  float* Qb      = take(NV);
  float* msg1    = take(NV);
  float* msgm    = take(NV);
  float* msgl    = take(NV);
  float* ffnt    = take((size_t)NTOT_ * 512);
  float* ffn1    = take((size_t)NTOT_ * 512);
  float* ffn2    = take(NV);
  float* qfea    = take(NV);
  float* k2b     = take(NV);
  float* v2b     = take(NV);
  float* Ksumb   = take(32 * 256);
  float* Zeb     = take(NTOT_);
  float* msg2    = take(NV);
  float* out2    = take(NV);
  float* g2b     = take(NV);

  // bf16 arena for packed (transposed) weights — 16B aligned (off is even).
  __bf16* bfa = (__bf16*)(wsf + off);
  size_t boff = 0;
  auto takeb = [&](size_t n) { __bf16* p = bfa + boff; boff += n; return p; };

  __bf16* ep2t = takeb(65536);
  __bf16* ep3t = takeb(65536);
  __bf16* ep4t = takeb(65536);
  __bf16* Awt  = takeb(2 * 65536);
  __bf16* Bwt  = takeb(2 * 65536);
  __bf16* Cwt  = takeb(2 * 65536);
  __bf16* Dwt  = takeb(2 * 65536);
  __bf16* Ewt  = takeb(2 * 65536);
  __bf16* wqt  = takeb(65536);
  __bf16* wkt  = takeb(65536);
  __bf16* wvt  = takeb(65536);
  __bf16* wmt  = takeb(65536);
  __bf16* w1t  = takeb(262144);   // [512][512]
  __bf16* w2t  = takeb(131072);   // [256][512]
  __bf16* KVt  = takeb((size_t)32 * 256 * 256);

  // ---- 0) one-time weight packing (f32 [K][N] -> bf16 [N][K]) ----
  auto pack = [&](const float* W, __bf16* dst, int K, int N) {
    packW<<<(K * N + 255) / 256, 256, 0, stream>>>(W, dst, K, N);
  };
  pack(ep2_w, ep2t, 256, 256);
  pack(ep3_w, ep3t, 256, 256);
  pack(ep4_w, ep4t, 256, 256);
  for (int l = 0; l < 2; ++l) {
    pack(A_w + l * 65536, Awt + l * 65536, 256, 256);
    pack(B_w + l * 65536, Bwt + l * 65536, 256, 256);
    pack(C_w + l * 65536, Cwt + l * 65536, 256, 256);
    pack(D_w + l * 65536, Dwt + l * 65536, 256, 256);
    pack(E_w + l * 65536, Ewt + l * 65536, 256, 256);
  }
  pack(wq, wqt, 256, 256);
  pack(wk, wkt, 256, 256);
  pack(wvv, wvt, 256, 256);
  pack(wm, wmt, 256, 256);
  pack(w1, w1t, 512, 512);
  pack(w2, w2t, 512, 256);

  // ---- 1) edge-local path: conv + ep2 GEMM ----
  conv_edge<<<ETOT_, 256, 0, stream>>>(h, esrc, edst, conv_w, conv_b, lrConv);
  gemm(stream, lrConv, ep2t, 256, ep2_b, nullptr, nullptr, lrLocal, ETOT_, 256, 256, 0);

  // ---- 2) hs = h @ ep3 + b ----
  gemm(stream, h, ep3t, 256, ep3_b, nullptr, nullptr, hsb, NTOT_, 256, 256, 0);

  // ---- 3) GatedGCN x2 (bn train stats, residual, relu) ----
  const float* hin = h;
  float* houts[2] = { hA, hBv };
  for (int l = 0; l < 2; ++l) {
    gemm(stream, hin, Awt + l * 65536, 256, A_b + l * 256, nullptr, nullptr, Ahb, NTOT_, 256, 256, 0);
    gemm(stream, hin, Bwt + l * 65536, 256, B_b + l * 256, nullptr, nullptr, Bhb, NTOT_, 256, 256, 0);
    gemm(stream, hin, Dwt + l * 65536, 256, D_b + l * 256, nullptr, nullptr, Dhb, NTOT_, 256, 256, 0);
    gemm(stream, hin, Ewt + l * 65536, 256, E_b + l * 256, nullptr, nullptr, Ehb, NTOT_, 256, 256, 0);
    gemm(stream, e,   Cwt + l * 65536, 256, C_b + l * 256, nullptr, nullptr, Ce,  ETOT_, 256, 256, 0);
    gcn_edge<<<32, 1024, 2 * 64 * 256 * sizeof(float), stream>>>(esrc, edst, Dhb, Ehb, Ce, Bhb, numb, denb);
    gcn_node<<<2048, 256, 0, stream>>>(Ahb, numb, denb, hn, (long long)NV);
    colstat_part<<<8, 256, 0, stream>>>(hn, NTOT_, part);
    colstat_fin<<<1, 256, 0, stream>>>(part, 8, NTOT_, meanv, varv);
    bn_apply<<<2048, 256, 0, stream>>>(hn, meanv, varv, bnh_g + l * 256, bnh_b + l * 256,
                                       hin, houts[l], (long long)NV);
    hin = houts[l];
  }

  // ---- 4) graph readout + encoder layer 1 (broadcast-source linear attn) ----
  mean_nodes<<<32, 256, 0, stream>>>(hBv, gg);
  gemm(stream, hsb, wqt, 256, nullptr, nullptr, nullptr, Qb,  NTOT_, 256, 256, 2);  // elu+1
  gemm(stream, gg,  wkt, 256, nullptr, nullptr, nullptr, Kgb, 32,    256, 256, 2);  // elu+1
  gemm(stream, gg,  wvt, 256, nullptr, nullptr, nullptr, vgb, 32,    256, 256, 0);
  attn1_kernel<<<256, 256, 0, stream>>>(Qb, Kgb, vgb, msg1, NTOT_);
  gemm(stream, msg1, wmt, 256, nullptr, nullptr, nullptr, msgm, NTOT_, 256, 256, 0);
  ln_kernel<<<256, 256, 0, stream>>>(msgm, ln1g, ln1b, nullptr, msgl, NTOT_);
  gemm(stream, hsb,  w1t,       512, nullptr, nullptr, nullptr, ffnt, NTOT_, 512, 256, 0);
  gemm(stream, msgl, w1t + 256, 512, nullptr, ffnt,    nullptr, ffn1, NTOT_, 512, 256, 1); // relu
  gemm(stream, ffn1, w2t,       512, nullptr, nullptr, nullptr, ffn2, NTOT_, 256, 512, 0);
  ln_kernel<<<256, 256, 0, stream>>>(ffn2, ln2g, ln2b, hsb, qfea, NTOT_);

  // ---- 5) encoder layer 2, collapsed to 2048 rows (tile/repeat identity) ----
  gemm(stream, qfea, wqt, 256, nullptr, nullptr, nullptr, Qb,  NTOT_, 256, 256, 2);  // Q2e
  gemm(stream, qfea, wkt, 256, nullptr, nullptr, nullptr, k2b, NTOT_, 256, 256, 0);
  gemm(stream, qfea, wvt, 256, nullptr, nullptr, nullptr, v2b, NTOT_, 256, 256, 0);
  kv_wmma<<<32, 256, 2 * 256 * 64 * sizeof(__bf16), stream>>>(k2b, v2b, KVt, Ksumb);
  z2_kernel<<<256, 256, 0, stream>>>(Qb, Ksumb, Zeb, NTOT_);
  gemm(stream, Qb, KVt, 256, nullptr, nullptr, Zeb, msg2, 64, 256, 256, 0, 4096.0f,
       32, 64 * 256, 256 * 256, 64 * 256, 64);                                  // batched Q@KV
  gemm(stream, msg2, wmt, 256, nullptr, nullptr, nullptr, msgm, NTOT_, 256, 256, 0);
  ln_kernel<<<256, 256, 0, stream>>>(msgm, ln1g, ln1b, nullptr, msgl, NTOT_);
  gemm(stream, qfea, w1t,       512, nullptr, nullptr, nullptr, ffnt, NTOT_, 512, 256, 0);
  gemm(stream, msgl, w1t + 256, 512, nullptr, ffnt,    nullptr, ffn1, NTOT_, 512, 256, 1);
  gemm(stream, ffn1, w2t,       512, nullptr, nullptr, nullptr, ffn2, NTOT_, 256, 512, 0);
  ln_kernel<<<256, 256, 0, stream>>>(ffn2, ln2g, ln2b, qfea, out2, NTOT_);

  // ---- 6) global edge features, fuse, final BatchNorm+ReLU ----
  gemm(stream, out2, ep4t, 256, ep4_b, nullptr, nullptr, g2b, NTOT_, 256, 256, 0);
  edge_pre<<<8192, 256, 0, stream>>>(lrLocal, g2b, edst, Ce, (long long)NE);
  colstat_part<<<256, 256, 0, stream>>>(Ce, ETOT_, part);
  colstat_fin<<<1, 256, 0, stream>>>(part, 256, ETOT_, meanv, varv);
  bn_apply<<<8192, 256, 0, stream>>>(Ce, meanv, varv, bn_g, bn_b, nullptr,
                                     (float*)d_out, (long long)NE);
}
